// RNN_15633680957749
// MI455X (gfx1250) — matrix operand order repaired and checked by reading, not compiled, not run on hardware
//
#include <hip/hip_runtime.h>

// ---------------------------------------------------------------------------
// Problem constants (from the reference)
// ---------------------------------------------------------------------------
#define TT 512
#define NN 128
#define INSZ 256
#define HH 512
#define NCLS 128

typedef __attribute__((ext_vector_type(16))) _Float16 v16h;
typedef __attribute__((ext_vector_type(8)))  _Float16 v8h;
typedef __attribute__((ext_vector_type(8)))  float    v8f;
typedef __attribute__((ext_vector_type(4)))  float    v4f;

// ---------------------------------------------------------------------------
// WMMA helper (CDNA5 v_wmma_f32_16x16x32_f16, wave32)
// ---------------------------------------------------------------------------
__device__ inline v8f wmma_f16(v16h a, v16h b, v8f c) {
  return __builtin_amdgcn_wmma_f32_16x16x32_f16(false, a, false, b, (short)0, c,
                                                false, false);
}

// Branch-free tanh: clamp then 1 - 2/(e^{2x}+1).  v_exp_f32 + v_rcp_f32,
// ~1 ulp of rcp accuracy, no EXEC divergence in the hot loop.
__device__ inline float fast_tanh(float x) {
  x = fminf(10.0f, fmaxf(-10.0f, x));
  float e = __expf(2.0f * x);
  return 1.0f - 2.0f * __builtin_amdgcn_rcpf(e + 1.0f);
}

// A-tile (16x32 f16) from row-major f32 source, converting on the fly.
// ISA 16-bit A layout: lanes 0-15 hold K = {0..7,16..23}, lanes 16-31 hold
// K = {8..15,24..31}; vector element e maps to K = base + (e<8 ? e : 8+e).
__device__ inline v16h load_a_f32(const float* __restrict__ A, int lda, int m,
                                  int k0, int lane) {
  const int hi = lane >> 4;
  const float* p = A + (size_t)m * lda + k0 + hi * 8;
  v4f x0 = *(const v4f*)(p);
  v4f x1 = *(const v4f*)(p + 4);
  v4f y0 = *(const v4f*)(p + 16);
  v4f y1 = *(const v4f*)(p + 20);
  v16h a;
#pragma unroll
  for (int i = 0; i < 4; ++i) {
    a[i]      = (_Float16)x0[i];
    a[4 + i]  = (_Float16)x1[i];
    a[8 + i]  = (_Float16)y0[i];
    a[12 + i] = (_Float16)y1[i];
  }
  return a;
}

// A-tile from row-major f16 source (two 16B vector loads per lane).
__device__ inline v16h load_a_f16(const _Float16* __restrict__ A, int lda,
                                  int m, int k0, int lane) {
  const int hi = lane >> 4;
  const _Float16* p = A + (size_t)m * lda + k0 + hi * 8;
  v8h lo = *(const v8h*)(p);
  v8h up = *(const v8h*)(p + 16);
  v16h a;
#pragma unroll
  for (int i = 0; i < 8; ++i) {
    a[i]     = lo[i];
    a[8 + i] = up[i];
  }
  return a;
}

// B operand from pre-packed weights: tile (kt, jt) stored as the exact WMMA
// per-lane register image -> one aligned 32-byte load per lane.
__device__ inline v16h load_b_packed(const _Float16* __restrict__ P,
                                     int jTiles, int kt, int jt, int lane) {
  const _Float16* p = P + ((size_t)(kt * jTiles + jt) << 9) + (lane << 4);
  return *(const v16h*)p;
}

// ---------------------------------------------------------------------------
// GEMM:  C[M,Ncols] = A[M,K] @ B[K,Ncols] + bias[Ncols]
// 128 threads = 4 waves / block; each wave computes a 16x128 tile.
// 4 waves/block leaves plenty of VGPR headroom, so B tiles are batched
// 4-at-a-time and the next A tile is prefetched under the WMMA group.
// grid = (M/64, Ncols/128).
// ---------------------------------------------------------------------------
template <bool AF32>
__global__ void __launch_bounds__(128)
gemm_bias_kernel(const void* __restrict__ Aptr,
                 const _Float16* __restrict__ Bp,
                 const float* __restrict__ bias, float* __restrict__ C, int M,
                 int Ncols, int K, int lda) {
  const int lane = threadIdx.x & 31;
  const int wave = threadIdx.x >> 5;
  const int mBase = blockIdx.x * 64 + wave * 16;
  const int jBase = blockIdx.y * 128;
  const int jt0 = jBase >> 4;
  const int jTiles = Ncols >> 4;
  const int m = mBase + (lane & 15);
  const int kTiles = K >> 5;

  v8f acc[8] = {};
  v16h a = AF32 ? load_a_f32((const float*)Aptr, lda, m, 0, lane)
                : load_a_f16((const _Float16*)Aptr, lda, m, 0, lane);
  for (int kt = 0; kt < kTiles; ++kt) {
    v16h b[4];
#pragma unroll
    for (int q = 0; q < 4; ++q)
      b[q] = load_b_packed(Bp, jTiles, kt, jt0 + q, lane);
#pragma unroll
    for (int q = 0; q < 4; ++q) acc[q] = wmma_f16(a, b[q], acc[q]);
#pragma unroll
    for (int q = 0; q < 4; ++q)
      b[q] = load_b_packed(Bp, jTiles, kt, jt0 + 4 + q, lane);
    v16h an = a;
    if (kt + 1 < kTiles)
      an = AF32 ? load_a_f32((const float*)Aptr, lda, m, (kt + 1) * 32, lane)
                : load_a_f16((const _Float16*)Aptr, lda, m, (kt + 1) * 32, lane);
#pragma unroll
    for (int q = 0; q < 4; ++q) acc[4 + q] = wmma_f16(a, b[q], acc[4 + q]);
    a = an;
  }

  const int col = lane & 15;
  const int hiL = lane >> 4;
#pragma unroll
  for (int q = 0; q < 8; ++q) {
    const int jj = jBase + q * 16 + col;
    const float bj = bias ? bias[jj] : 0.0f;
#pragma unroll
    for (int r = 0; r < 8; ++r) {
      const int mm = mBase + r + hiL * 8;
      C[(size_t)mm * Ncols + jj] = acc[q][r] + bj;
    }
  }
}

// ---------------------------------------------------------------------------
// Persistent recurrent layer: one 1024-thread workgroup, h (128x512 f32)
// resident in LDS (256 KB of the WGP's 320 KB).  Per step:
//   h_new = tanh(xproj[t] + h @ Whh^T);  h = mask ? h_new : h
// 32 waves: wave w -> rows [16*(w>>2)), cols [128*(w&3)) (8 WMMA tiles each).
//
// NOTE on register budget: 1024 threads = 8 waves/SIMD, so each wave gets
// only ~128 VGPRs.  The K-loop therefore keeps a minimal live set (acc[8]=64
// + one A + one B) and relies on the 8 resident waves per SIMD for latency
// hiding.  '#pragma unroll 1' stops the compiler from fully unrolling the
// 16-iteration loop and spilling B tiles to scratch (seen in round 3).
// ---------------------------------------------------------------------------
__global__ void __launch_bounds__(1024)
rnn_layer_kernel(const float* __restrict__ h0, const float* __restrict__ xproj,
                 const _Float16* __restrict__ WhhP,
                 const int* __restrict__ lengths, _Float16* __restrict__ outSeq,
                 float* __restrict__ hFinal, int nSteps) {
  extern __shared__ float hS[];  // NN * HH floats = 256 KB

  const int tid = threadIdx.x;
  const int lane = tid & 31;
  const int wave = tid >> 5;
  const int mBase = (wave >> 2) * 16;
  const int jBase = (wave & 3) * 128;
  const int jt0 = jBase >> 4;
  const int m = mBase + (lane & 15);
  const int col = lane & 15;
  const int hiL = lane >> 4;
  const int jTiles = HH >> 4;  // 32

  // Load initial hidden state into LDS (vectorized).
  for (int i = tid; i < NN * HH / 4; i += 1024)
    ((v4f*)hS)[i] = ((const v4f*)h0)[i];

  // Per-lane sequence lengths for the 8 rows this lane updates.
  int lenR[8];
#pragma unroll
  for (int r = 0; r < 8; ++r) lenR[r] = lengths[mBase + r + hiL * 8];

  // This lane owns elements (mBase + hiL*8 + r, jBase + q*16 + col).
  const size_t updBase = (size_t)(mBase + hiL * 8) * HH + jBase + col;

  __syncthreads();

  for (int t = 0; t < nSteps; ++t) {
    const float* __restrict__ xp = xproj + (size_t)t * (NN * HH);

    // Prefetch next step's xproj slab (256 KB) into cache.
    if (t + 1 < nSteps) {
      const char* nxt = (const char*)(xproj + (size_t)(t + 1) * (NN * HH));
      __builtin_prefetch(nxt + (size_t)tid * 256, 0, 1);
    }

    v8f acc[8] = {};
#pragma unroll 1
    for (int kt = 0; kt < 16; ++kt) {
      v16h a = load_a_f32(hS, HH, m, kt * 32, lane);  // LDS, f32 -> f16
#pragma unroll
      for (int q = 0; q < 8; ++q) {
        v16h b = load_b_packed(WhhP, jTiles, kt, jt0 + q, lane);
        acc[q] = wmma_f16(a, b, acc[q]);
      }
    }

    __syncthreads();  // all reads of hS done before anyone writes

    _Float16* __restrict__ os =
        outSeq ? outSeq + (size_t)t * (NN * HH) : (_Float16*)0;
#pragma unroll
    for (int q = 0; q < 8; ++q) {
      const size_t qBase = updBase + q * 16;
      // Batch the 8 xproj loads + 8 old-h LDS reads so they issue as
      // clauses with a single wait, instead of load/wait/add per element.
      float xv[8], ov[8];
#pragma unroll
      for (int r = 0; r < 8; ++r) xv[r] = xp[qBase + (size_t)r * HH];
#pragma unroll
      for (int r = 0; r < 8; ++r) ov[r] = hS[qBase + (size_t)r * HH];
#pragma unroll
      for (int r = 0; r < 8; ++r) {
        const float hn = fast_tanh(acc[q][r] + xv[r]);
        const float hv = (t < lenR[r]) ? hn : ov[r];
        hS[qBase + (size_t)r * HH] = hv;
        if (os) os[qBase + (size_t)r * HH] = (_Float16)hv;
      }
    }

    __syncthreads();  // writes visible before next step's reads
  }

  // Emit final hidden state (f32) straight into its d_out slot.
  for (int i = tid; i < NN * HH / 4; i += 1024)
    ((v4f*)hFinal)[i] = ((const v4f*)hS)[i];
}

// ---------------------------------------------------------------------------
// Weight prep: W[J][Kc] (f32 row-major, out_features x in_features) ->
// packed f16 B tiles in the WMMA per-lane register image.
//   B[k][j] = W[j][k];  tile (kt, jt):
//   packed[(kt*jTiles + jt)*512 + lane*16 + e] = B[kt*32 + (lane>=16)*8 + (e<8 ? e : 8+e)]
//                                                 [jt*16 + (lane&15)]
// ---------------------------------------------------------------------------
__global__ void pack_b_f16(const float* __restrict__ W,
                           _Float16* __restrict__ P, int J, int Kc) {
  const int idx = blockIdx.x * blockDim.x + threadIdx.x;
  if (idx >= J * Kc) return;
  const int jTiles = J >> 4;
  const int e = idx & 15;
  const int lane = (idx >> 4) & 31;
  const int tile = idx >> 9;
  const int jt = tile % jTiles;
  const int kt = tile / jTiles;
  const int k = kt * 32 + ((lane >> 4) << 3) + (e < 8 ? e : 8 + e);
  const int j = jt * 16 + (lane & 15);
  P[idx] = (_Float16)W[(size_t)j * Kc + k];
}

__global__ void add_bias_kernel(const float* __restrict__ a,
                                const float* __restrict__ b,
                                float* __restrict__ o, int n) {
  int i = blockIdx.x * blockDim.x + threadIdx.x;
  if (i < n) o[i] = a[i] + b[i];
}

// ---------------------------------------------------------------------------
// Host launcher
// ---------------------------------------------------------------------------
extern "C" void kernel_launch(void* const* d_in, const int* in_sizes, int n_in,
                              void* d_out, int out_size, void* d_ws,
                              size_t ws_size, hipStream_t stream) {
  (void)in_sizes; (void)n_in; (void)out_size; (void)ws_size;

  const float* x       = (const float*)d_in[0];   // (T, N, 256)
  const float* h0      = (const float*)d_in[1];   // (2, N, 512)
  const int*   lengths = (const int*)d_in[2];     // (N,)
  const float* Wih0    = (const float*)d_in[3];   // (512, 256)
  const float* bih0    = (const float*)d_in[4];
  const float* Whh0    = (const float*)d_in[5];   // (512, 512)
  const float* bhh0    = (const float*)d_in[6];
  const float* Wih1    = (const float*)d_in[7];   // (512, 512)
  const float* bih1    = (const float*)d_in[8];
  const float* Whh1    = (const float*)d_in[9];   // (512, 512)
  const float* bhh1    = (const float*)d_in[10];
  const float* Wfc     = (const float*)d_in[11];  // (128, 512)
  const float* bfc     = (const float*)d_in[12];

  float* out = (float*)d_out;             // logits(128*128) | h0f | h1f
  float* out_h0 = out + NCLS * NN;        // offset 16384
  float* out_h1 = out_h0 + NN * HH;       // offset 81920

  // Workspace carve-up (256-byte aligned).
  char* ws = (char*)d_ws;
  size_t off = 0;
  auto carve = [&](size_t bytes) -> void* {
    void* p = ws + off;
    off = (off + bytes + 255) & ~(size_t)255;
    return p;
  };
  float*    xproj = (float*)carve((size_t)TT * NN * HH * sizeof(float));  // reused by both layers
  _Float16* out0h = (_Float16*)carve((size_t)TT * NN * HH * sizeof(_Float16));
  _Float16* Wih0P = (_Float16*)carve((size_t)INSZ * HH * sizeof(_Float16));
  _Float16* Whh0P = (_Float16*)carve((size_t)HH * HH * sizeof(_Float16));
  _Float16* Wih1P = (_Float16*)carve((size_t)HH * HH * sizeof(_Float16));
  _Float16* Whh1P = (_Float16*)carve((size_t)HH * HH * sizeof(_Float16));
  _Float16* WfcP  = (_Float16*)carve((size_t)HH * NCLS * sizeof(_Float16));
  float*    b0    = (float*)carve(HH * sizeof(float));
  float*    b1    = (float*)carve(HH * sizeof(float));

  // --- weight prep (f32 -> WMMA-packed f16 tiles) + bias folding ---
  pack_b_f16<<<(HH * INSZ + 255) / 256, 256, 0, stream>>>(Wih0, Wih0P, HH, INSZ);
  pack_b_f16<<<(HH * HH + 255) / 256, 256, 0, stream>>>(Whh0, Whh0P, HH, HH);
  pack_b_f16<<<(HH * HH + 255) / 256, 256, 0, stream>>>(Wih1, Wih1P, HH, HH);
  pack_b_f16<<<(HH * HH + 255) / 256, 256, 0, stream>>>(Whh1, Whh1P, HH, HH);
  pack_b_f16<<<(NCLS * HH + 255) / 256, 256, 0, stream>>>(Wfc, WfcP, NCLS, HH);
  add_bias_kernel<<<(HH + 255) / 256, 256, 0, stream>>>(bih0, bhh0, b0, HH);
  add_bias_kernel<<<(HH + 255) / 256, 256, 0, stream>>>(bih1, bhh1, b1, HH);

  const size_t ldsBytes = (size_t)NN * HH * sizeof(float);  // 256 KB

  // --- layer 0: xproj0 = x @ Wih0^T + (bih0+bhh0), then recurrence ---
  dim3 gX(TT * NN / 64, HH / 128);  // (1024, 4)
  gemm_bias_kernel<true><<<gX, 128, 0, stream>>>(x, Wih0P, b0, xproj,
                                                 TT * NN, HH, INSZ, INSZ);
  rnn_layer_kernel<<<1, 1024, ldsBytes, stream>>>(h0, xproj, Whh0P, lengths,
                                                  out0h, out_h0, TT);

  // --- layer 1: xproj1 = out0 @ Wih1^T + (bih1+bhh1), then recurrence ---
  gemm_bias_kernel<false><<<gX, 128, 0, stream>>>(out0h, Wih1P, b1, xproj,
                                                  TT * NN, HH, HH, HH);
  rnn_layer_kernel<<<1, 1024, ldsBytes, stream>>>(h0 + NN * HH, xproj, Whh1P,
                                                  lengths, (_Float16*)0,
                                                  out_h1, TT);

  // --- logits = h1_final @ Wfc^T + bfc  (mask freeze => last == final h) ---
  dim3 gF(NN / 64, NCLS / 128);  // (2, 1)
  gemm_bias_kernel<true><<<gF, 128, 0, stream>>>(out_h1, WfcP, bfc, out,
                                                 NN, NCLS, HH, HH);
}
